// SNNLinear_42219528519840
// MI455X (gfx1250) — matrix operand order repaired and checked by reading, not compiled, not run on hardware
//
#include <hip/hip_runtime.h>

typedef __attribute__((ext_vector_type(16))) _Float16 v16h;
typedef __attribute__((ext_vector_type(8)))  _Float16 v8h;
typedef __attribute__((ext_vector_type(8)))  float    v8f;

#define T_STEPS 100
#define BATCH_N 64
#define IN_F    1024
#define OUT_F   1024
#define THR_V   15.0f
#define TCHUNK  25

// d_ws layout:
//   [0)                : spikes as f16, [T*B, IN] row-major        = 13,107,200 B
//   [SP16_BYTES)       : W panels f16, [64 o-tiles][32 kc][512]    =  2,097,152 B
#define SP16_HALFS   ((size_t)T_STEPS * BATCH_N * IN_F)
#define SP16_BYTES   (SP16_HALFS * 2)
#define WP_HALFS_PER_TILE (32 * 512)

// ---- pack spikes f32 -> f16 (plain row-major; 0/1 values are exact in f16) ----
__global__ __launch_bounds__(256) void pack_spikes_kernel(
    const float* __restrict__ spikes, _Float16* __restrict__ sp16)
{
    const size_t i4 = ((size_t)blockIdx.x * 256 + threadIdx.x) * 4;
    const float4 v = *(const float4*)(spikes + i4);
    _Float16* d = sp16 + i4;
    d[0] = (_Float16)v.x; d[1] = (_Float16)v.y;
    d[2] = (_Float16)v.z; d[3] = (_Float16)v.w;
}

// ---- pack W f32 -> f16 into WMMA-B (KxN) lane-layout panels per o-tile ----
// Element W[o][k]: ot=o/16, n=o%16, kc=k/32, kk=k%32,
//   lane = n + 16*(kk/16), slot = kk%16
//   dst[(ot*32 + kc)*512 + lane*16 + slot]
__global__ __launch_bounds__(256) void pack_w_kernel(
    const float* __restrict__ W, _Float16* __restrict__ wp)
{
    const int idx = blockIdx.x * 256 + threadIdx.x;   // 0 .. 1M-1
    const int o = idx >> 10;
    const int k = idx & (IN_F - 1);
    const int ot = o >> 4, n = o & 15;
    const int kc = k >> 5, kk = k & 31;
    const int bl = n + ((kk >> 4) << 4);
    const int sl = kk & 15;
    wp[(size_t)(ot * 32 + kc) * 512 + bl * 16 + sl] = (_Float16)W[idx];
}

// ---- fused GEMM (WMMA) + LIF scan ----
__global__ __launch_bounds__(256) void snn_fused_kernel(
    const _Float16* __restrict__ sp16,  // [T*B, IN] f16
    const _Float16* __restrict__ wp,    // packed W panels
    const float* __restrict__ mem0,     // [B, OUT]
    const float* __restrict__ bias,     // [OUT]
    float* __restrict__ out)            // ss | mem_out | hat_s
{
    __shared__ __align__(32) _Float16 Bs[32 * 512];      // 32 KB, identical layout to ws panel
    __shared__ __align__(16) float    Ds[TCHUNK * 256];  // 25.6 KB

    const int b0 = blockIdx.x * 16;
    const int ot = blockIdx.y;
    const int n0 = ot * 16;
    const int tid  = threadIdx.x;
    const int wave = tid >> 5;
    const int lane = tid & 31;
    const int m    = lane & 15;
    const int hi   = lane >> 4;

    // ---- stage this o-tile's 32KB W panel into LDS with async global->LDS copies ----
    {
        const _Float16* src = wp + (size_t)ot * WP_HALFS_PER_TILE;
        #pragma unroll
        for (int j = 0; j < 8; ++j) {
            const int chunk = tid + j * 256;            // 16B chunks, 2048 total
            const unsigned           ldsa = (unsigned)(unsigned long long)(Bs + chunk * 8);
            const unsigned long long ga   = (unsigned long long)(src + chunk * 8);
            asm volatile("global_load_async_to_lds_b128 %0, %1, off"
                         :: "v"(ldsa), "v"(ga) : "memory");
        }
        asm volatile("s_wait_asynccnt 0" ::: "memory");
    }
    __syncthreads();

    // per-thread scan state: thread tid owns tile element (M = tid/16, N = tid%16)
    const int bb = b0 + (tid >> 4);
    const int oo = n0 + (tid & 15);
    float       mval = mem0[bb * OUT_F + oo];
    float       ssum = 0.0f;
    const float bi   = bias[oo];

    const _Float16* arow_base = sp16 + (size_t)(b0 + m) * IN_F;

    for (int c = 0; c < T_STEPS / TCHUNK; ++c) {
        // ---- GEMM phase: waves round-robin over timesteps of this chunk ----
        for (int tt = wave; tt < TCHUNK; tt += 8) {
            const int t = c * TCHUNK + tt;
            const _Float16* arow = arow_base + (size_t)t * (BATCH_N * IN_F);
            v8f acc = {};
            for (int kc = 0; kc < IN_F / 32; ++kc) {
                // A layout: lane(hi=0) K 0..15 ; lane(hi=1) K 16..31
                const v8h alo = *(const v8h*)(arow + kc * 32 + hi * 16);
                const v8h ahi = *(const v8h*)(arow + kc * 32 + hi * 16 + 8);
                const v16h a = __builtin_shufflevector(alo, ahi,
                        0, 1, 2, 3, 4, 5, 6, 7, 8, 9, 10, 11, 12, 13, 14, 15);
                const v16h bmat = *(const v16h*)(Bs + kc * 512 + lane * 16);
                acc = __builtin_amdgcn_wmma_f32_16x16x32_f16(
                        false, a, false, bmat, (short)0, acc, false, false);
            }
            // C/D layout: VGPR r, lane L -> M = r + 8*(L/16), N = L%16
            float* drow = Ds + tt * 256;
            #pragma unroll
            for (int r = 0; r < 8; ++r)
                drow[(r + 8 * hi) * 16 + m] = acc[r];
        }
        __syncthreads();

        // ---- scan phase: 25 LIF steps on this thread's tile element ----
        for (int tt = 0; tt < TCHUNK; ++tt) {
            const int t = c * TCHUNK + tt;
            const float du = Ds[tt * 256 + tid] + bi;
            mval += du;
            const float s = (mval > THR_V) ? 1.0f : 0.0f;
            mval = fminf(fmaxf(mval, 0.0f), THR_V);
            mval -= mval * s;
            ssum += s;
            out[(size_t)(t * BATCH_N + bb) * OUT_F + oo] = s;
        }
        __syncthreads();
    }

    const size_t base = (size_t)T_STEPS * BATCH_N * OUT_F;
    out[base + bb * OUT_F + oo] = mval;
    out[base + BATCH_N * OUT_F + bb * OUT_F + oo] = ssum * (1.0f / T_STEPS);
}

extern "C" void kernel_launch(void* const* d_in, const int* in_sizes, int n_in,
                              void* d_out, int out_size, void* d_ws, size_t ws_size,
                              hipStream_t stream) {
    const float* spikes = (const float*)d_in[0];   // [100, 64, 1024]
    const float* mem0   = (const float*)d_in[1];   // [64, 1024]
    // d_in[2] = hat_spikes: numerically irrelevant to forward outputs
    const float* W      = (const float*)d_in[3];   // [1024, 1024]
    const float* bias   = (const float*)d_in[4];   // [1024]
    float* out = (float*)d_out;
    (void)in_sizes; (void)n_in; (void)out_size; (void)ws_size;

    _Float16* sp16 = (_Float16*)d_ws;
    _Float16* wp   = (_Float16*)((char*)d_ws + SP16_BYTES);

    // pack spikes: 6,553,600 elems, 4 per thread -> 6400 blocks
    pack_spikes_kernel<<<dim3((unsigned)(SP16_HALFS / 4 / 256)), dim3(256), 0, stream>>>(spikes, sp16);
    // pack W: 1,048,576 elems, 1 per thread -> 4096 blocks
    pack_w_kernel<<<dim3(OUT_F * IN_F / 256), dim3(256), 0, stream>>>(W, wp);

    dim3 grid(BATCH_N / 16, OUT_F / 16);   // 4 x 64 = 256 blocks
    snn_fused_kernel<<<grid, dim3(256), 0, stream>>>(sp16, wp, mem0, bias, out);
}
